// ActorNetwork_3470333575770
// MI455X (gfx1250) — matrix operand order repaired and checked by reading, not compiled, not run on hardware
//
#include <hip/hip_runtime.h>
#include <hip/hip_bf16.h>

typedef __attribute__((ext_vector_type(16))) _Float16 v16h;
typedef __attribute__((ext_vector_type(8)))  _Float16 v8h;
typedef __attribute__((ext_vector_type(8)))  float    v8f;

#define WPB 8            // waves per block (wave32)
#define TPB (WPB * 32)

struct MLP3 { const float *W0, *b0, *W1, *b1, *W2, *b2; };

// ---------------------------------------------------------------------------
// tanh: gfx1250 has v_tanh_f32 (TRANS op, co-executes with WMMA)
// ---------------------------------------------------------------------------
__device__ __forceinline__ float fast_tanh(float x) {
#if __has_builtin(__builtin_amdgcn_tanhf)
  return __builtin_amdgcn_tanhf(x);
#elif __has_builtin(__builtin_amdgcn_tanh_f32)
  return __builtin_amdgcn_tanh_f32(x);
#else
  float x2 = x * x;
  float p = x * (135135.0f + x2 * (17325.0f + x2 * (378.0f + x2)));
  float q = 135135.0f + x2 * (62370.0f + x2 * (3150.0f + 28.0f * x2));
  return fminf(1.0f, fmaxf(-1.0f, p / q));
#endif
}

// ---------------------------------------------------------------------------
// WMMA fragment helpers (ISA 7.12.2 layouts), all LDS-resident, vectorized
// ---------------------------------------------------------------------------
__device__ __forceinline__ v8f wmma16(v16h a, v16h b, v8f c) {
  return __builtin_amdgcn_wmma_f32_16x16x32_f16(false, a, false, b, (short)0, c,
                                                false, false);
}

__device__ __forceinline__ v16h cat8(v8h lo, v8h hi) {
  return __builtin_shufflevector(lo, hi, 0, 1, 2, 3, 4, 5, 6, 7,
                                 8, 9, 10, 11, 12, 13, 14, 15);
}

// A fragment 16x32: lane=M, halves 0..7 = K[kg..kg+8), halves 8..15 = K[16+kg..)
// -> two contiguous 16B LDS loads per lane.
__device__ __forceinline__ v16h load_A_frag(const _Float16* As, int stride,
                                            int kofs, int lane) {
  int m = lane & 15;
  int kg = (lane >> 4) * 8;
  const _Float16* p = As + m * stride + kofs + kg;
  return cat8(*(const v8h*)p, *(const v8h*)(p + 16));
}

// B fragment 32x16 from zero-padded f16 LDS panel (row stride ldn halves):
// lane&15 = N col, halves 0..7 = K[kg..kg+8), halves 8..15 = K[16+kg..).
__device__ __forceinline__ v16h load_B_frag(const _Float16* Wl, int ldn,
                                            int kofs, int nbase, int lane) {
  int n = lane & 15;
  int kg = (lane >> 4) * 8;
  const _Float16* p = Wl + (kofs + kg) * ldn + nbase + n;
  v16h b;
#pragma unroll
  for (int s = 0; s < 8; ++s) {
    b[s] = p[s * ldn];
    b[8 + s] = p[(16 + s) * ldn];
  }
  return b;
}

__device__ __forceinline__ v8f mlp_tile(const _Float16* Ain, int sin,
                                        const _Float16* Wl, int ldn, int nKt,
                                        int nbase, int lane) {
  v8f c = {};
  for (int kt = 0; kt < nKt; ++kt)
    c = wmma16(load_A_frag(Ain, sin, kt * 32, lane),
               load_B_frag(Wl, ldn, kt * 32, nbase, lane), c);
  return c;
}

// D tile -> LDS f16 with bias (+opt tanh). lane&15 = N col, vgpr j -> M row.
__device__ __forceinline__ void store_D_lds(v8f d, _Float16* Aout, int stride,
                                            int nbase, const float* bias,
                                            bool act, int lane) {
  int n = lane & 15;
  int mg = (lane >> 4) * 8;
  float bb = bias[nbase + n];
#pragma unroll
  for (int j = 0; j < 8; ++j) {
    float v = d[j] + bb;
    if (act) v = fast_tanh(v);
    Aout[(mg + j) * stride + nbase + n] = (_Float16)v;
  }
}

__device__ __forceinline__ void store_D_global_f32(v8f d, float* out, int base,
                                                   int rows, const float* bias,
                                                   int lane) {
  int n = lane & 15;
  int mg = (lane >> 4) * 8;
  float bb = bias[n];
#pragma unroll
  for (int j = 0; j < 8; ++j) {
    int r = base + mg + j;
    if (r < rows) out[(long long)r * 16 + n] = d[j] + bb;
  }
}

__device__ __forceinline__ void mlp_layer(const _Float16* Ain, int sin,
                                          _Float16* Aout, int sout,
                                          const _Float16* Wl, int ldn, int nKt,
                                          const float* bias, int N, bool act,
                                          int lane) {
  for (int nb = 0; nb < N; nb += 16) {
    v8f c = mlp_tile(Ain, sin, Wl, ldn, nKt, nb, lane);
    store_D_lds(c, Aout, sout, nb, bias, act, lane);
  }
}

// ---------------------------------------------------------------------------
// Block-level weight staging: zero-padded f16 panels in LDS (shared by 8 waves)
//   W0: 64x32 (K<=64, N=32)   W1: 32x16 (K=32)   W2: 32x16 (K<=16, N<=16)
// ---------------------------------------------------------------------------
__device__ __forceinline__ void stage3(MLP3 M, int K0, int N0, int K2, int N2,
                                       _Float16* cW0, _Float16* cW1,
                                       _Float16* cW2, float* cb0, float* cb1,
                                       float* cb2, int tid) {
  for (int i = tid; i < 64 * 32; i += TPB) {
    int k = i >> 5, n = i & 31;
    cW0[i] = (k < K0 && n < N0) ? (_Float16)M.W0[k * N0 + n] : (_Float16)0.0f;
  }
  for (int i = tid; i < 32 * 16; i += TPB) {
    int k = i >> 4, n = i & 15;
    cW1[i] = (k < 32) ? (_Float16)M.W1[k * 16 + n] : (_Float16)0.0f;
    cW2[i] = (k < K2 && n < N2) ? (_Float16)M.W2[k * N2 + n] : (_Float16)0.0f;
  }
  if (tid < 32)      cb0[tid] = M.b0[tid];
  else if (tid < 48) cb1[tid - 32] = M.b1[tid - 32];
  else if (tid < 64) cb2[tid - 48] = (tid - 48 < N2) ? M.b2[tid - 48] : 0.0f;
}

#define DECL_WCACHE(tag)                                            \
  __shared__ alignas(16) _Float16 tag##W0[64 * 32];                 \
  __shared__ alignas(16) _Float16 tag##W1[32 * 16];                 \
  __shared__ alignas(16) _Float16 tag##W2[32 * 16];                 \
  __shared__ float tag##b0[32], tag##b1[16], tag##b2[16];

// Per-wave LDS scratch: in(16x64) | t1(16x32) | t2(16x32) = 2048 halves = 4KB
#define SCR_HALVES 2048
#define DECL_SCRATCH()                                              \
  __shared__ alignas(16) _Float16 lds_[WPB][SCR_HALVES];            \
  int lane = threadIdx.x & 31;                                      \
  int wid  = threadIdx.x >> 5;                                      \
  _Float16* in = lds_[wid];                                         \
  _Float16* t1 = in + 1024;                                         \
  _Float16* t2 = in + 1536;                                         \
  for (int i = lane; i < SCR_HALVES; i += 32) in[i] = (_Float16)0.0f;

__device__ __forceinline__ void store_row16_h(_Float16* dst,
                                              const _Float16* srcLds) {
  *(v8h*)dst = *(const v8h*)srcLds;
  *(v8h*)(dst + 8) = *(const v8h*)(srcLds + 8);
}

// ---------------------------------------------------------------------------
// Kernels
// ---------------------------------------------------------------------------
__global__ void k_zero(float* p, long long n) {
  long long t = blockIdx.x * (long long)blockDim.x + threadIdx.x;
  long long stride = (long long)gridDim.x * blockDim.x;
  for (; t < n; t += stride) p[t] = 0.0f;
}

// prep MLP (2->32->16->16) then proc MLP (16->32->16->16); store f16
__global__ void k_prep_proc(const float* __restrict__ x, int N, MLP3 P, MLP3 Q,
                            _Float16* __restrict__ xprep,
                            _Float16* __restrict__ xproc) {
  DECL_WCACHE(p)
  DECL_WCACHE(q)
  DECL_SCRATCH();
  stage3(P, 2, 32, 16, 16, pW0, pW1, pW2, pb0, pb1, pb2, threadIdx.x);
  stage3(Q, 16, 32, 16, 16, qW0, qW1, qW2, qb0, qb1, qb2, threadIdx.x);
  __syncthreads();
  int tile = blockIdx.x * WPB + wid;
  if (tile >= (N + 15) / 16) return;
  int base = tile * 16;
  if (lane < 16) {
    int node = base + lane;
    if (node < N) {
      in[lane * 64 + 0] = (_Float16)x[node * 5 + 3];
      in[lane * 64 + 1] = (_Float16)x[node * 5 + 4];
    }
  }
  // prep
  mlp_layer(in, 64, t1, 32, pW0, 32, 1, pb0, 32, true, lane);
  mlp_layer(t1, 32, t2, 32, pW1, 16, 1, pb1, 16, true, lane);
  {
    v8f c = mlp_tile(t2, 32, pW2, 16, 1, 0, lane);
    store_D_lds(c, in, 64, 0, pb2, false, lane);  // x_prep -> 'in' cols 0..15
  }
  if (lane < 16) {
    int node = base + lane;
    if (node < N) store_row16_h(xprep + (long long)node * 16, in + lane * 64);
  }
  // proc (reads x_prep from 'in'; cols 16..31 still zero)
  mlp_layer(in, 64, t1, 32, qW0, 32, 1, qb0, 32, true, lane);
  mlp_layer(t1, 32, t2, 32, qW1, 16, 1, qb1, 16, true, lane);
  {
    v8f c = mlp_tile(t2, 32, qW2, 16, 1, 0, lane);
    store_D_lds(c, t1, 32, 0, qb2, false, lane);
  }
  if (lane < 16) {
    int node = base + lane;
    if (node < N) store_row16_h(xproc + (long long)node * 16, t1 + lane * 32);
  }
}

// edge scatter: agg[dst] += x_proc[src]  (dim-parallel, f32 atomics)
__global__ void k_edge(const int* __restrict__ src, const int* __restrict__ dst,
                       const _Float16* __restrict__ xproc,
                       float* __restrict__ agg, long long E) {
  long long t = blockIdx.x * (long long)blockDim.x + threadIdx.x;
  if (t >= E * 16) return;
  int dim = (int)(t & 15);
  long long e = t >> 4;
  int s = src[e], d = dst[e];
  atomicAdd(&agg[(long long)d * 16 + dim],
            (float)xproc[(long long)s * 16 + dim]);
}

// agg MLP (16->32->16->16), node_emb = x_prep + out; store f16
__global__ void k_aggmlp(const float* __restrict__ agg,
                         const _Float16* __restrict__ xprep, int N, MLP3 A,
                         _Float16* __restrict__ nemb) {
  DECL_WCACHE(a)
  DECL_SCRATCH();
  stage3(A, 16, 32, 16, 16, aW0, aW1, aW2, ab0, ab1, ab2, threadIdx.x);
  __syncthreads();
  int tile = blockIdx.x * WPB + wid;
  if (tile >= (N + 15) / 16) return;
  int base = tile * 16;
  if (lane < 16) {
    int node = base + lane;
    if (node < N) {
      const float4* src = (const float4*)(agg + (long long)node * 16);
#pragma unroll
      for (int q = 0; q < 4; ++q) {
        float4 v = src[q];
        in[lane * 64 + q * 4 + 0] = (_Float16)v.x;
        in[lane * 64 + q * 4 + 1] = (_Float16)v.y;
        in[lane * 64 + q * 4 + 2] = (_Float16)v.z;
        in[lane * 64 + q * 4 + 3] = (_Float16)v.w;
      }
    }
  }
  mlp_layer(in, 64, t1, 32, aW0, 32, 1, ab0, 32, true, lane);
  mlp_layer(t1, 32, t2, 32, aW1, 16, 1, ab1, 16, true, lane);
  v8f c = mlp_tile(t2, 32, aW2, 16, 1, 0, lane);
  store_D_lds(c, t1, 32, 0, ab2, false, lane);
  if (lane < 16) {
    int node = base + lane;
    if (node < N) {
      v8h h0 = *(const v8h*)(t1 + lane * 32);
      v8h h1 = *(const v8h*)(t1 + lane * 32 + 8);
      v8h x0 = *(const v8h*)(xprep + (long long)node * 16);
      v8h x1 = *(const v8h*)(xprep + (long long)node * 16 + 8);
      v8h r0, r1;
#pragma unroll
      for (int q = 0; q < 8; ++q) {
        r0[q] = (_Float16)((float)h0[q] + (float)x0[q]);
        r1[q] = (_Float16)((float)h1[q] + (float)x1[q]);
      }
      *(v8h*)(nemb + (long long)node * 16) = r0;
      *(v8h*)(nemb + (long long)node * 16 + 8) = r1;
    }
  }
}

// per-DAG sum of node_comb = [node_feat(2), node_emb(16)]  (contiguous nodes)
__global__ void k_ncagg(const float* __restrict__ x,
                        const _Float16* __restrict__ nemb,
                        float* __restrict__ ncagg, int G, int npd) {
  int t = blockIdx.x * blockDim.x + threadIdx.x;
  if (t >= G * 18) return;
  int g = t / 18, d = t % 18;
  long long base = (long long)g * npd;
  float s = 0.0f;
  if (d < 2) {
    for (int i = 0; i < npd; ++i) s += x[(base + i) * 5 + 3 + d];
  } else {
    int dd = d - 2;
    for (int i = 0; i < npd; ++i) s += (float)nemb[(base + i) * 16 + dd];
  }
  ncagg[(long long)g * 18 + d] = s;
}

// dag MLP (20->32->16->16) -> dag_emb f32
__global__ void k_dagmlp(const float* __restrict__ x, const int* __restrict__ ptr,
                         const float* __restrict__ ncagg, int G, MLP3 D,
                         float* __restrict__ demb) {
  DECL_WCACHE(d)
  DECL_SCRATCH();
  stage3(D, 20, 32, 16, 16, dW0, dW1, dW2, db0, db1, db2, threadIdx.x);
  __syncthreads();
  int tile = blockIdx.x * WPB + wid;
  if (tile >= (G + 15) / 16) return;
  int base = tile * 16;
  if (lane < 16) {
    int g = base + lane;
    if (g < G) {
      long long p = ptr[g];
      in[lane * 64 + 0] = (_Float16)x[p * 5 + 1];
      in[lane * 64 + 1] = (_Float16)x[p * 5 + 2];
      for (int k = 0; k < 18; ++k)
        in[lane * 64 + 2 + k] = (_Float16)ncagg[(long long)g * 18 + k];
    }
  }
  mlp_layer(in, 64, t1, 32, dW0, 32, 1, db0, 32, true, lane);
  mlp_layer(t1, 32, t2, 32, dW1, 16, 1, db1, 16, true, lane);
  v8f c = mlp_tile(t2, 32, dW2, 16, 1, 0, lane);
  store_D_global_f32(c, demb, base, G, db2, lane);
}

// per-obs sum of dag_emb (contiguous dags)
__global__ void k_dagagg(const float* __restrict__ demb, float* __restrict__ dagg,
                         int B, int dpo) {
  int t = blockIdx.x * blockDim.x + threadIdx.x;
  if (t >= B * 16) return;
  int o = t >> 4, d = t & 15;
  long long base = (long long)o * dpo;
  float s = 0.0f;
  for (int i = 0; i < dpo; ++i) s += demb[(base + i) * 16 + d];
  dagg[(long long)o * 16 + d] = s;
}

// obs_indptr (cumsum of num_dags_per_obs) and num_nodes_per_obs -> d_out tail
__global__ void k_obsmeta(const int* __restrict__ ptr, const int* __restrict__ ndpo,
                          int B, float* __restrict__ out_nn,
                          float* __restrict__ out_ip) {
  int t = blockIdx.x * blockDim.x + threadIdx.x;
  if (t > B) return;
  int s = 0;
  for (int i = 0; i < t; ++i) s += ndpo[i];
  out_ip[t] = (float)s;
  if (t < B) {
    int e = s + ndpo[t];
    out_nn[t] = (float)(ptr[e] - ptr[s]);  // telescoping per-dag node counts
  }
}

// glob MLP (17->32->16->16) -> global_emb f32
__global__ void k_globmlp(const float* __restrict__ x,
                          const float* __restrict__ out_ip,
                          const float* __restrict__ dagg, int B, MLP3 GP,
                          float* __restrict__ gemb) {
  DECL_WCACHE(g)
  DECL_SCRATCH();
  stage3(GP, 17, 32, 16, 16, gW0, gW1, gW2, gb0, gb1, gb2, threadIdx.x);
  __syncthreads();
  int tile = blockIdx.x * WPB + wid;
  if (tile >= (B + 15) / 16) return;
  int base = tile * 16;
  if (lane < 16) {
    int o = base + lane;
    if (o < B) {
      int idx = (int)out_ip[o];
      in[lane * 64 + 0] = (_Float16)x[(long long)idx * 5 + 0];
      const float4* src = (const float4*)(dagg + (long long)o * 16);
#pragma unroll
      for (int q = 0; q < 4; ++q) {
        float4 v = src[q];
        in[lane * 64 + 1 + q * 4 + 0] = (_Float16)v.x;
        in[lane * 64 + 1 + q * 4 + 1] = (_Float16)v.y;
        in[lane * 64 + 1 + q * 4 + 2] = (_Float16)v.z;
        in[lane * 64 + 1 + q * 4 + 3] = (_Float16)v.w;
      }
    }
  }
  mlp_layer(in, 64, t1, 32, gW0, 32, 1, gb0, 32, true, lane);
  mlp_layer(t1, 32, t2, 32, gW1, 16, 1, gb1, 16, true, lane);
  v8f c = mlp_tile(t2, 32, gW2, 16, 1, 0, lane);
  store_D_global_f32(c, gemb, base, B, gb2, lane);
}

// node_score MLP (48->32->16->1); final layer as 16-wide VALU dot
__global__ void k_nodescore(const _Float16* __restrict__ nemb,
                            const float* __restrict__ demb,
                            const float* __restrict__ gemb,
                            const int* __restrict__ batch, int N, int dpo,
                            MLP3 S, float* __restrict__ out) {
  DECL_WCACHE(s)
  DECL_SCRATCH();
  stage3(S, 48, 32, 16, 1, sW0, sW1, sW2, sb0, sb1, sb2, threadIdx.x);
  __syncthreads();
  int tile = blockIdx.x * WPB + wid;
  if (tile >= (N + 15) / 16) return;
  int base = tile * 16;
  if (lane < 16) {
    int node = base + lane;
    if (node < N) {
      int g = batch[node];
      int o = g / dpo;
      *(v8h*)(in + lane * 64) = *(const v8h*)(nemb + (long long)node * 16);
      *(v8h*)(in + lane * 64 + 8) =
          *(const v8h*)(nemb + (long long)node * 16 + 8);
      const float4* dg = (const float4*)(demb + (long long)g * 16);
      const float4* gg = (const float4*)(gemb + (long long)o * 16);
#pragma unroll
      for (int q = 0; q < 4; ++q) {
        float4 dv = dg[q], gv = gg[q];
        in[lane * 64 + 16 + q * 4 + 0] = (_Float16)dv.x;
        in[lane * 64 + 16 + q * 4 + 1] = (_Float16)dv.y;
        in[lane * 64 + 16 + q * 4 + 2] = (_Float16)dv.z;
        in[lane * 64 + 16 + q * 4 + 3] = (_Float16)dv.w;
        in[lane * 64 + 32 + q * 4 + 0] = (_Float16)gv.x;
        in[lane * 64 + 32 + q * 4 + 1] = (_Float16)gv.y;
        in[lane * 64 + 32 + q * 4 + 2] = (_Float16)gv.z;
        in[lane * 64 + 32 + q * 4 + 3] = (_Float16)gv.w;
      }
    }
  }
  mlp_layer(in, 64, t1, 32, sW0, 32, 2, sb0, 32, true, lane);  // K=48 -> 2 tiles
  mlp_layer(t1, 32, t2, 32, sW1, 16, 1, sb1, 16, true, lane);
  if (lane < 16) {
    int node = base + lane;
    if (node < N) {
      float s = sb2[0];
#pragma unroll
      for (int k = 0; k < 16; ++k)
        s += (float)t2[lane * 32 + k] * (float)sW2[k * 16];
      out[node] = s;
    }
  }
}

// dag_score MLP (33->32->16->1) over G*W rows
__global__ void k_dagscore(const float* __restrict__ demb,
                           const float* __restrict__ gemb, int G, int W, int dpo,
                           MLP3 S, float* __restrict__ out) {
  DECL_WCACHE(s)
  DECL_SCRATCH();
  stage3(S, 33, 32, 16, 1, sW0, sW1, sW2, sb0, sb1, sb2, threadIdx.x);
  __syncthreads();
  long long R = (long long)G * W;
  long long tile = (long long)blockIdx.x * WPB + wid;
  if (tile >= (R + 15) / 16) return;
  long long base = tile * 16;
  if (lane < 16) {
    long long r = base + lane;
    if (r < R) {
      int g = (int)(r / W), w = (int)(r % W);
      int o = g / dpo;
      const float4* dg = (const float4*)(demb + (long long)g * 16);
      const float4* gg = (const float4*)(gemb + (long long)o * 16);
#pragma unroll
      for (int q = 0; q < 4; ++q) {
        float4 dv = dg[q], gv = gg[q];
        in[lane * 64 + q * 4 + 0] = (_Float16)dv.x;
        in[lane * 64 + q * 4 + 1] = (_Float16)dv.y;
        in[lane * 64 + q * 4 + 2] = (_Float16)dv.z;
        in[lane * 64 + q * 4 + 3] = (_Float16)dv.w;
        in[lane * 64 + 16 + q * 4 + 0] = (_Float16)gv.x;
        in[lane * 64 + 16 + q * 4 + 1] = (_Float16)gv.y;
        in[lane * 64 + 16 + q * 4 + 2] = (_Float16)gv.z;
        in[lane * 64 + 16 + q * 4 + 3] = (_Float16)gv.w;
      }
      in[lane * 64 + 32] = (_Float16)(float)w;
    }
  }
  mlp_layer(in, 64, t1, 32, sW0, 32, 2, sb0, 32, true, lane);  // K=33 -> 2 tiles
  mlp_layer(t1, 32, t2, 32, sW1, 16, 1, sb1, 16, true, lane);
  if (lane < 16) {
    long long r = base + lane;
    if (r < R) {
      float s = sb2[0];
#pragma unroll
      for (int k = 0; k < 16; ++k)
        s += (float)t2[lane * 32 + k] * (float)sW2[k * 16];
      out[r] = s;
    }
  }
}

// ---------------------------------------------------------------------------
extern "C" void kernel_launch(void* const* d_in, const int* in_sizes, int n_in,
                              void* d_out, int out_size, void* d_ws,
                              size_t ws_size, hipStream_t stream) {
  const float* x   = (const float*)d_in[0];
  const int* ei    = (const int*)d_in[1];
  const int* ptr   = (const int*)d_in[2];
  const int* batch = (const int*)d_in[3];
  const int* ndpo  = (const int*)d_in[4];

  int N = in_sizes[0] / 5;
  long long E = in_sizes[1] / 2;
  int G = in_sizes[2] - 1;
  int B = in_sizes[4];
  int W = (int)(((long long)out_size - N - 2LL * B - 1) / G);  // num_workers
  int npd = N / G;  // nodes per dag
  int dpo = G / B;  // dags per obs

  // params flattened in JAX pytree (sorted-key) order:
  // agg, dag, dag_score, glob, node_score, prep, proc — each (W0,b0,W1,b1,W2,b2)
  auto mlp = [&](int i) {
    MLP3 m{(const float*)d_in[i],     (const float*)d_in[i + 1],
           (const float*)d_in[i + 2], (const float*)d_in[i + 3],
           (const float*)d_in[i + 4], (const float*)d_in[i + 5]};
    return m;
  };
  MLP3 agg = mlp(6), dag = mlp(12), dsc = mlp(18), glb = mlp(24),
       nsc = mlp(30), prep = mlp(36), proc = mlp(42);

  // workspace layout (256B aligned slabs)
  char* ws = (char*)d_ws;
  size_t off = 0;
  auto alloc = [&](size_t bytes) {
    void* p = ws + off;
    off = (off + bytes + 255) & ~(size_t)255;
    return p;
  };
  _Float16* xprep = (_Float16*)alloc((size_t)N * 16 * 2);
  _Float16* xproc = (_Float16*)alloc((size_t)N * 16 * 2);
  float*    aggb  = (float*)   alloc((size_t)N * 16 * 4);
  _Float16* nemb  = (_Float16*)alloc((size_t)N * 16 * 2);
  float*    ncagg = (float*)   alloc((size_t)G * 18 * 4);
  float*    demb  = (float*)   alloc((size_t)G * 16 * 4);
  float*    dagg  = (float*)   alloc((size_t)B * 16 * 4);
  float*    gemb  = (float*)   alloc((size_t)B * 16 * 4);
  (void)ws_size; (void)n_in;

  float* out = (float*)d_out;
  long long R = (long long)G * W;
  float* out_nn = out + (size_t)N + (size_t)R;  // num_nodes_per_obs
  float* out_ip = out_nn + B;                   // obs_indptr

  int tilesN = (N + 15) / 16;
  int tilesG = (G + 15) / 16;
  int tilesB = (B + 15) / 16;
  long long tilesR = (R + 15) / 16;

  k_zero<<<2048, 256, 0, stream>>>(aggb, (long long)N * 16);
  k_prep_proc<<<(tilesN + WPB - 1) / WPB, TPB, 0, stream>>>(x, N, prep, proc,
                                                            xprep, xproc);
  long long etot = E * 16;
  k_edge<<<(int)((etot + 255) / 256), 256, 0, stream>>>(ei, ei + E, xproc, aggb, E);
  k_aggmlp<<<(tilesN + WPB - 1) / WPB, TPB, 0, stream>>>(aggb, xprep, N, agg, nemb);
  k_ncagg<<<(G * 18 + 255) / 256, 256, 0, stream>>>(x, nemb, ncagg, G, npd);
  k_dagmlp<<<(tilesG + WPB - 1) / WPB, TPB, 0, stream>>>(x, ptr, ncagg, G, dag, demb);
  k_dagagg<<<(B * 16 + 255) / 256, 256, 0, stream>>>(demb, dagg, B, dpo);
  k_obsmeta<<<(B + 1 + 255) / 256, 256, 0, stream>>>(ptr, ndpo, B, out_nn, out_ip);
  k_globmlp<<<(tilesB + WPB - 1) / WPB, TPB, 0, stream>>>(x, out_ip, dagg, B, glb, gemb);
  k_nodescore<<<(tilesN + WPB - 1) / WPB, TPB, 0, stream>>>(nemb, demb, gemb, batch,
                                                            N, dpo, nsc, out);
  k_dagscore<<<(int)((tilesR + WPB - 1) / WPB), TPB, 0, stream>>>(demb, gemb, G, W,
                                                                  dpo, dsc, out + N);
}
